// LocalAttentionBlock_25391846654026
// MI455X (gfx1250) — matrix operand order repaired and checked by reading, not compiled, not run on hardware
//
#include <hip/hip_runtime.h>
#include <cstdint>
#include <cstddef>

typedef _Float16 half_t;
typedef __attribute__((ext_vector_type(16))) _Float16 v16h;
typedef __attribute__((ext_vector_type(8)))  _Float16 v8h;
typedef __attribute__((ext_vector_type(8)))  float    v8f;
typedef __attribute__((ext_vector_type(4)))  unsigned int u32x4;
typedef __attribute__((ext_vector_type(4)))  int          i32x4;
typedef __attribute__((ext_vector_type(8)))  int          i32x8;

union V16 { v16h v; v8h h[2]; };

static constexpr int kB   = 2;
static constexpr int kT   = 4096;
static constexpr int kW   = 2048;
static constexpr int kNH  = 16;
static constexpr int kHD  = 128;
static constexpr int kWIN = 1024;
static constexpr int kM   = kB * kT;     // 8192 rows total

#if defined(__gfx1250__) && __has_builtin(__builtin_amdgcn_tensor_load_to_lds)
#define USE_TDM 1
#else
#define USE_TDM 0
#endif

__device__ __forceinline__ v8f wmma_f16(v16h a, v16h b, v8f c) {
  return __builtin_amdgcn_wmma_f32_16x16x32_f16(false, a, false, b, (short)0, c,
                                                false, false);
}

#if USE_TDM
// Issue one TDM 2-D tile load (f16 elements) into LDS.
// tile0 = elements per row, tile1 = rows, stride0 = tensor row stride (elems).
__device__ __forceinline__ void tdm_load_2d(uint32_t lds_off, uint64_t gaddr,
                                            uint32_t tile0, uint32_t tile1,
                                            uint64_t stride0) {
  const uint32_t td0 = 0x00ffffffu;   // huge tensor dims: no OOB clipping
  const uint32_t td1 = 0x00ffffffu;
  u32x4 g0;
  g0[0] = 1u;                                               // count=1, user D#
  g0[1] = lds_off;                                          // LDS byte addr
  g0[2] = (uint32_t)gaddr;                                  // global addr lo
  g0[3] = ((uint32_t)(gaddr >> 32) & 0x01ffffffu) | 0x80000000u; // hi | type=2
  i32x8 g1;
  g1[0] = (int)(1u << 16);                                  // data_size=2B
  g1[1] = (int)((td0 & 0xffffu) << 16);                     // tensor_dim0 lo
  g1[2] = (int)((td0 >> 16) | ((td1 & 0xffffu) << 16));     // d0 hi | d1 lo
  g1[3] = (int)((td1 >> 16) | (tile0 << 16));               // d1 hi | tile0
  g1[4] = (int)(tile1 & 0xffffu);                           // tile1 | tile2=0
  g1[5] = (int)(uint32_t)stride0;                           // stride0[31:0]
  g1[6] = (int)((uint32_t)(stride0 >> 32) & 0xffffu);       // stride0[47:32]
  g1[7] = 0;
  i32x4 gz = {0, 0, 0, 0};
#if defined(__clang_major__) && (__clang_major__ >= 23)
  i32x8 gz8 = {0, 0, 0, 0, 0, 0, 0, 0};
  __builtin_amdgcn_tensor_load_to_lds(g0, g1, gz, gz, gz8, 0);
#else
  __builtin_amdgcn_tensor_load_to_lds(g0, g1, gz, gz, 0);
#endif
}
#endif

// ---------------------------------------------------------------------------
// f32 -> f16 conversion (straight)
// ---------------------------------------------------------------------------
__global__ void cvt_f32_f16_25391846654026(const float* __restrict__ src,
                                           half_t* __restrict__ dst, size_t n) {
  size_t i = (size_t)blockIdx.x * blockDim.x + threadIdx.x;
  if (i < n) dst[i] = (half_t)src[i];
}

// f32 [K][N] -> f16 [N][K] (transpose, so B-fragments become contiguous)
__global__ void cvtT_f32_f16_25391846654026(const float* __restrict__ src,
                                            half_t* __restrict__ dst,
                                            int K, int N) {
  size_t i = (size_t)blockIdx.x * blockDim.x + threadIdx.x;
  size_t total = (size_t)K * N;
  if (i >= total) return;
  int k = (int)(i / N);
  int n = (int)(i % N);
  dst[(size_t)n * K + k] = (half_t)src[i];
}

// ---------------------------------------------------------------------------
// Partial RoPE: rotate pairs (i, i+32), i in [0,32), angle = pos*10000^(-i/32)
// x layout: [B, T, heads, 128] f16, in-place.
// ---------------------------------------------------------------------------
__global__ void rope_f16_25391846654026(half_t* __restrict__ x,
                                        const int* __restrict__ segpos,
                                        int heads) {
  size_t idx = (size_t)blockIdx.x * blockDim.x + threadIdx.x;
  size_t total = (size_t)kB * kT * heads * 32;
  if (idx >= total) return;
  int i = (int)(idx & 31);
  size_t tmp = idx >> 5;
  int h = (int)(tmp % heads); tmp /= heads;
  int t = (int)(tmp % kT);
  int b = (int)(tmp / kT);
  int pos = segpos[(size_t)b * kT + t];
  float inv_ts = powf(10000.0f, -((float)i / 32.0f));
  float ang = (float)pos * inv_ts;
  float sn = sinf(ang), cs = cosf(ang);
  half_t* base = x + (((size_t)((size_t)b * kT + t) * heads + h) * kHD);
  float first  = (float)base[i];
  float second = (float)base[i + 32];
  base[i]      = (half_t)(first * cs - second * sn);
  base[i + 32] = (half_t)(second * cs + first * sn);
}

// ---------------------------------------------------------------------------
// WMMA GEMM: C[M,N] = A[M,K] (row-major f16) * Bt[N,K] (col-major f16).
// 8 waves / block; each wave computes a 16 x (16*TN) strip (A reused TN times).
// STORE_T=1 writes the output transposed ([N][M]) for V.
// ---------------------------------------------------------------------------
template <int TN, int STORE_T>
__global__ void __launch_bounds__(256, 1)
gemm_wmma_25391846654026(const half_t* __restrict__ A,
                         const half_t* __restrict__ Bt,
                         const float* __restrict__ bias,
                         half_t* __restrict__ outH,
                         float* __restrict__ outF,
                         int M, int N, int K) {
  const int wave = threadIdx.x >> 5;
  const int lane = threadIdx.x & 31;
  const int col  = lane & 15;
  const int hi   = (lane >= 16) ? 1 : 0;
  const int n0   = (blockIdx.x * 8 + wave) * 16 * TN;
  const int m0   = blockIdx.y * 16;

  const v8f vzero = {0.f, 0.f, 0.f, 0.f, 0.f, 0.f, 0.f, 0.f};
  v8f acc[TN];
#pragma unroll
  for (int tn = 0; tn < TN; ++tn) acc[tn] = vzero;

  const half_t* arow = A + (size_t)(m0 + col) * K;
  for (int k0 = 0; k0 < K; k0 += 32) {
    if (k0 + 256 < K) {
      __builtin_prefetch(arow + k0 + 256, 0, 1);
      __builtin_prefetch(&Bt[(size_t)(n0 + col) * K + k0 + 256], 0, 1);
    }
    V16 a;
    const v8h* pA = (const v8h*)(arow + k0 + hi * 16);
    a.h[0] = pA[0];              // K [base, base+8)
    a.h[1] = pA[1];              // K [base+8, base+16)
#pragma unroll
    for (int tn = 0; tn < TN; ++tn) {
      v16h b = *(const v16h*)&Bt[(size_t)(n0 + tn * 16 + col) * K + k0 + hi * 16];
      acc[tn] = wmma_f16(a.v, b, acc[tn]);
    }
  }

  const int rbase = m0 + hi * 8;
#pragma unroll
  for (int tn = 0; tn < TN; ++tn) {
    const int nc = n0 + tn * 16 + col;
#pragma unroll
    for (int r = 0; r < 8; ++r) {
      float v = acc[tn][r];
      if (bias) v += bias[nc];
      if (STORE_T) {
        outH[(size_t)nc * M + rbase + r] = (half_t)v;          // [N][M]
      } else if (outF) {
        outF[(size_t)(rbase + r) * N + nc] = v;
      } else {
        outH[(size_t)(rbase + r) * N + nc] = (half_t)v;
      }
    }
  }
}

// ---------------------------------------------------------------------------
// Flash attention, MQA, causal window 1024.
// Block = 8 waves = 8 heads sharing one 16-query tile. K/V chunks staged into
// double-buffered LDS; with TDM the next chunk's DMA overlaps the current
// chunk's 16 WMMAs. Interior key-steps (fully inside the causal window) take
// a mask-free fast path; only the window-edge and diagonal steps do compares.
// qh: [B,T,NH,HD] f16 (RoPE'd)   kh: [B,T,HD] f16 (RoPE'd)
// vtg: [HD][M] f16 (transposed)  eh: [B,T,NH*HD] f16
// ---------------------------------------------------------------------------
__global__ void __launch_bounds__(256, 1)
attn_wmma_25391846654026(const half_t* __restrict__ qh,
                         const half_t* __restrict__ kh,
                         const half_t* __restrict__ vtg,
                         half_t* __restrict__ eh) {
  __shared__ _Float16 Klds[2][32 * kHD];    // ping-pong [key][128]  2 x 8 KB
  __shared__ _Float16 VTlds[2][kHD * 32];   // ping-pong [c][key]    2 x 8 KB
  __shared__ _Float16 Plds[8 * 16 * 32];    // per-wave P tile       8 KB

  const int tid  = threadIdx.x;
  const int wave = tid >> 5;
  const int lane = tid & 31;
  const int t0   = blockIdx.x * 16;
  const int head = blockIdx.y * 8 + wave;
  const int bidx = blockIdx.z;
  const int col  = lane & 15;
  const int hi   = (lane >= 16) ? 1 : 0;
  const int rbase = hi * 8;
  const int boff  = bidx * kT;

  const half_t* kbB = kh + (size_t)bidx * kT * kHD;

  // Q A-fragments: 16x128 as 4 chunks of K=32, vector-loaded, kept live.
  V16 qa[4];
  const half_t* qrow =
      qh + ((size_t)((size_t)bidx * kT + t0 + col) * kNH + head) * kHD;
#pragma unroll
  for (int cix = 0; cix < 4; ++cix) {
    const v8h* p = (const v8h*)(qrow + cix * 32 + hi * 16);
    qa[cix].h[0] = p[0];
    qa[cix].h[1] = p[1];
  }

  const v8f vzero = {0.f, 0.f, 0.f, 0.f, 0.f, 0.f, 0.f, 0.f};
  v8f O[8];
#pragma unroll
  for (int i = 0; i < 8; ++i) O[i] = vzero;
  float m_i[8], l_i[8];
#pragma unroll
  for (int r = 0; r < 8; ++r) { m_i[r] = -1.0e4f; l_i[r] = 0.0f; }

  int s_begin = t0 - (kWIN - 1);
  if (s_begin < 0) s_begin = 0;
  s_begin &= ~31;
  const int s_end = t0 + 16;
  const float scale = 0.088388347648318447f;  // 1/sqrt(128)

#if USE_TDM
  if (wave == 0) {  // pre-issue first chunk
    tdm_load_2d((uint32_t)(size_t)&Klds[0][0],
                (uint64_t)(size_t)(kbB + (size_t)s_begin * kHD),
                4096u, 1u, 4096u);
    tdm_load_2d((uint32_t)(size_t)&VTlds[0][0],
                (uint64_t)(size_t)(vtg + (size_t)boff + s_begin),
                32u, 128u, (uint64_t)kM);
  }
#endif

  int ib = 0;
  for (int s0 = s_begin; s0 < s_end; s0 += 32) {
#if USE_TDM
    if (wave == 0) __builtin_amdgcn_s_wait_tensorcnt(0);
    __syncthreads();  // chunk ib ready for everyone
    if (wave == 0 && s0 + 32 < s_end) {  // prefetch next chunk under compute
      tdm_load_2d((uint32_t)(size_t)&Klds[ib ^ 1][0],
                  (uint64_t)(size_t)(kbB + (size_t)(s0 + 32) * kHD),
                  4096u, 1u, 4096u);
      tdm_load_2d((uint32_t)(size_t)&VTlds[ib ^ 1][0],
                  (uint64_t)(size_t)(vtg + (size_t)boff + s0 + 32),
                  32u, 128u, (uint64_t)kM);
    }
#else
    __syncthreads();  // previous readers done before restaging
    {  // cooperative staging: 256 threads x 32 B each per chunk
      *(v16h*)&Klds[ib][tid * 16] =
          *(const v16h*)&kbB[(size_t)s0 * kHD + tid * 16];
      const int c = tid >> 1, j0 = (tid & 1) * 16;
      *(v16h*)&VTlds[ib][c * 32 + j0] =
          *(const v16h*)&vtg[(size_t)c * kM + boff + s0 + j0];
    }
    __syncthreads();
#endif

    // ---- logits: S = q . k^T * scale; mask-free fast path when the whole
    //      32-key block is inside the causal window for all 16 rows ----
    const bool full = (s0 + 31 <= t0) && (s0 >= t0 - (kWIN - 16));
    float Sv[2][8];
    if (full) {
#pragma unroll
      for (int sub = 0; sub < 2; ++sub) {
        v8f c8 = vzero;
        const int keyl = sub * 16 + col;
#pragma unroll
        for (int cix = 0; cix < 4; ++cix) {
          v16h bfr = *(const v16h*)&Klds[ib][keyl * kHD + cix * 32 + hi * 16];
          c8 = wmma_f16(qa[cix].v, bfr, c8);
        }
#pragma unroll
        for (int r = 0; r < 8; ++r) Sv[sub][r] = c8[r] * scale;
      }
    } else {
#pragma unroll
      for (int sub = 0; sub < 2; ++sub) {
        v8f c8 = vzero;
        const int keyl = sub * 16 + col;
#pragma unroll
        for (int cix = 0; cix < 4; ++cix) {
          v16h bfr = *(const v16h*)&Klds[ib][keyl * kHD + cix * 32 + hi * 16];
          c8 = wmma_f16(qa[cix].v, bfr, c8);
        }
#pragma unroll
        for (int r = 0; r < 8; ++r) {
          int rg = t0 + rbase + r;
          int sg = s0 + keyl;
          bool ok = (sg <= rg) && ((rg - sg) < kWIN);
          Sv[sub][r] = ok ? (c8[r] * scale) : -30000.0f;
        }
      }
    }

    // ---- online softmax (rows live in 16-lane half-wave groups) ----
    float alpha[8];
    half_t p0[8], p1[8];
#pragma unroll
    for (int r = 0; r < 8; ++r) {
      float v = fmaxf(Sv[0][r], Sv[1][r]);
      v = fmaxf(v, __shfl_xor(v, 1, 16));
      v = fmaxf(v, __shfl_xor(v, 2, 16));
      v = fmaxf(v, __shfl_xor(v, 4, 16));
      v = fmaxf(v, __shfl_xor(v, 8, 16));
      float mnew = fmaxf(m_i[r], v);
      alpha[r] = __expf(m_i[r] - mnew);
      m_i[r] = mnew;
      float e0 = __expf(Sv[0][r] - mnew);
      float e1 = __expf(Sv[1][r] - mnew);
      p0[r] = (half_t)e0;
      p1[r] = (half_t)e1;
      float s = e0 + e1;
      s += __shfl_xor(s, 1, 16);
      s += __shfl_xor(s, 2, 16);
      s += __shfl_xor(s, 4, 16);
      s += __shfl_xor(s, 8, 16);
      l_i[r] = l_i[r] * alpha[r] + s;
    }
#pragma unroll
    for (int n = 0; n < 8; ++n)
#pragma unroll
      for (int r = 0; r < 8; ++r) O[n][r] *= alpha[r];

    // ---- reshape P (C/D layout) -> A fragment via per-wave LDS tile ----
    half_t* pw = &Plds[wave * 512];
#pragma unroll
    for (int r = 0; r < 8; ++r) {
      pw[(rbase + r) * 32 + col]      = p0[r];
      pw[(rbase + r) * 32 + 16 + col] = p1[r];
    }
    __syncthreads();
    V16 pa;
    {
      const half_t* prow = pw + col * 32;
      pa.h[0] = *(const v8h*)(prow + hi * 16);
      pa.h[1] = *(const v8h*)(prow + hi * 16 + 8);
    }

    // ---- O += P(16x32) * V(32x128), V^T fragments contiguous in LDS ----
#pragma unroll
    for (int n = 0; n < 8; ++n) {
      v16h vfr = *(const v16h*)&VTlds[ib][(n * 16 + col) * 32 + hi * 16];
      O[n] = wmma_f16(pa.v, vfr, O[n]);
    }

    __syncthreads();  // all readers done with chunk ib before it is restaged
    ib ^= 1;
  }

  // ---- finalize: O * (1/l), write encoded [B,T,NH*HD] ----
  float inv_l[8];
#pragma unroll
  for (int r = 0; r < 8; ++r) inv_l[r] = __frcp_rn(l_i[r] + 1e-20f);
#pragma unroll
  for (int n = 0; n < 8; ++n) {
#pragma unroll
    for (int r = 0; r < 8; ++r) {
      int rg = t0 + rbase + r;
      float o = O[n][r] * inv_l[r];
      eh[((size_t)((size_t)bidx * kT + rg) * kW) + head * kHD + n * 16 + col] =
          (half_t)o;
    }
  }
}

// ---------------------------------------------------------------------------
// Host launcher
// ---------------------------------------------------------------------------
extern "C" void kernel_launch(void* const* d_in, const int* in_sizes, int n_in,
                              void* d_out, int out_size, void* d_ws,
                              size_t ws_size, hipStream_t stream) {
  (void)in_sizes; (void)n_in; (void)out_size; (void)ws_size;
  const float* x  = (const float*)d_in[0];
  const float* Wq = (const float*)d_in[1];
  const float* Wk = (const float*)d_in[2];
  const float* Wv = (const float*)d_in[3];
  const float* Wf = (const float*)d_in[4];
  const float* bf = (const float*)d_in[5];
  const int*   sp = (const int*)d_in[6];
  float* out = (float*)d_out;

  char* ws = (char*)d_ws;
  const size_t nX  = (size_t)kM * kW;   // 16,777,216
  const size_t nWq = (size_t)kW * kW;   //  4,194,304
  const size_t nWk = (size_t)kW * kHD;  //    262,144

  half_t* xh   = (half_t*)(ws + 0);          // 33,554,432 B (reused as encoded)
  half_t* WqT  = (half_t*)(ws + 33554432);   //  8,388,608 B  [N=2048][K=2048]
  half_t* WkT  = (half_t*)(ws + 41943040);   //    524,288 B  [N=128][K=2048]
  half_t* WvT  = (half_t*)(ws + 42467328);   //    524,288 B
  half_t* WfT  = (half_t*)(ws + 42991616);   //  8,388,608 B
  half_t* qh   = (half_t*)(ws + 51380224);   // 33,554,432 B  [M][2048]
  half_t* kkh  = (half_t*)(ws + 84934656);   //  2,097,152 B  [M][128]
  half_t* vT   = (half_t*)(ws + 87031808);   //  2,097,152 B  [128][M]
  half_t* eh   = xh;

  dim3 b256(256);
  cvt_f32_f16_25391846654026<<<(unsigned)((nX + 255) / 256), b256, 0, stream>>>(
      x, xh, nX);
  cvtT_f32_f16_25391846654026<<<(unsigned)((nWq + 255) / 256), b256, 0, stream>>>(
      Wq, WqT, kW, kW);
  cvtT_f32_f16_25391846654026<<<(unsigned)((nWk + 255) / 256), b256, 0, stream>>>(
      Wk, WkT, kW, kHD);
  cvtT_f32_f16_25391846654026<<<(unsigned)((nWk + 255) / 256), b256, 0, stream>>>(
      Wv, WvT, kW, kHD);
  cvtT_f32_f16_25391846654026<<<(unsigned)((nWq + 255) / 256), b256, 0, stream>>>(
      Wf, WfT, kW, kW);

  // Q: [M,2048] = xh @ Wq      (TN=2: wave strip 16x32)
  gemm_wmma_25391846654026<2, 0><<<dim3(kW / 256, kM / 16), b256, 0, stream>>>(
      xh, WqT, nullptr, qh, nullptr, kM, kW, kW);
  // K: [M,128] = xh @ Wk       (TN=1)
  gemm_wmma_25391846654026<1, 0><<<dim3(1, kM / 16), b256, 0, stream>>>(
      xh, WkT, nullptr, kkh, nullptr, kM, kHD, kW);
  // V^T: [128,M] = (xh @ Wv)^T (TN=1, transposed store)
  gemm_wmma_25391846654026<1, 1><<<dim3(1, kM / 16), b256, 0, stream>>>(
      xh, WvT, nullptr, vT, nullptr, kM, kHD, kW);

  const size_t nRopeQ = (size_t)kB * kT * kNH * 32;
  const size_t nRopeK = (size_t)kB * kT * 1 * 32;
  rope_f16_25391846654026<<<(unsigned)((nRopeQ + 255) / 256), b256, 0, stream>>>(
      qh, sp, kNH);
  rope_f16_25391846654026<<<(unsigned)((nRopeK + 255) / 256), b256, 0, stream>>>(
      kkh, sp, 1);

  attn_wmma_25391846654026<<<dim3(kT / 16, kNH / 8, kB), b256, 0, stream>>>(
      qh, kkh, vT, eh);

  // out: [M,2048] f32 = eh @ Wf + bf
  gemm_wmma_25391846654026<2, 0><<<dim3(kW / 256, kM / 16), b256, 0, stream>>>(
      eh, WfT, bf, nullptr, out, kM, kW, kW);
}
